// UMambaBlock_26405458936117
// MI455X (gfx1250) — matrix operand order repaired and checked by reading, not compiled, not run on hardware
//
#include <hip/hip_runtime.h>

typedef __bf16 bf16_t;
typedef __attribute__((ext_vector_type(16))) __bf16 v16bf;
typedef __attribute__((ext_vector_type(8)))  __bf16 v8bf;
typedef __attribute__((ext_vector_type(8)))  float  v8f;

#define HH   24
#define HP   26         // padded spatial extent (1-voxel zero halo)
#define LL   13824      // 24^3
#define LP   17576      // 26^3
#define BB   2
#define BLM  27648      // BB*LL  (divisible by 16)
#define DIMC 64
#define DIN  128
#define NST  16

// ---------------------------------------------------------------------------
// Pre-kernels: repack / convert to bf16 in WMMA-friendly layouts
// ---------------------------------------------------------------------------

// x [B, C, 24^3] f32 -> xt [B, 26,26,26, C] bf16 (channel-fastest, zero halo)
__global__ void umamba_pad_transpose(const float* __restrict__ x, bf16_t* __restrict__ xt) {
    long idx = (long)blockIdx.x * blockDim.x + threadIdx.x;
    long n = (long)BB * LP * DIMC;
    if (idx >= n) return;
    int c = idx % DIMC; long r = idx / DIMC;
    int px = r % HP; r /= HP;
    int py = r % HP; r /= HP;
    int pz = r % HP; int b = (int)(r / HP);
    float v = 0.f;
    if (px >= 1 && px < 25 && py >= 1 && py < 25 && pz >= 1 && pz < 25)
        v = x[((long)b * DIMC + c) * LL + (pz - 1) * (HH * HH) + (py - 1) * HH + (px - 1)];
    xt[idx] = (bf16_t)v;
}

// w [O, I, 27] f32 -> wt [27, O, I] bf16
__global__ void umamba_repack_convw(const float* __restrict__ w, bf16_t* __restrict__ wt,
                                    int O, int I) {
    long idx = (long)blockIdx.x * blockDim.x + threadIdx.x;
    long n = (long)O * I * 27;
    if (idx >= n) return;
    int tap = idx % 27; long oi = idx / 27; int i = oi % I; int o = oi / I;
    wt[((long)tap * O + o) * I + i] = (bf16_t)w[idx];
}

// plain f32 -> bf16 copy (weights already [N, K] k-fastest)
__global__ void umamba_f2bf(const float* __restrict__ in, bf16_t* __restrict__ out, long n) {
    long idx = (long)blockIdx.x * blockDim.x + threadIdx.x;
    if (idx < n) out[idx] = (bf16_t)in[idx];
}

// ---------------------------------------------------------------------------
// WMMA fragment helpers (wave32; A/B use contiguous k order, C/D as in ISA 7.12.2)
// A 16x32 bf16: lane row = lane&15; elem j -> K = 16*laneg + j
// B 32x16 bf16: lane col = lane&15; elem j -> K = 16*laneg + j
// C/D f32 16x16: lane holds (M = 8*laneg + v, N = lane&15)
// ---------------------------------------------------------------------------
__device__ inline v16bf umamba_load_a(const bf16_t* p /* &row[k_chunk + 16*laneg] */) {
    v8bf lo = *(const v8bf*)p;
    v8bf hi = *(const v8bf*)(p + 8);
    v16bf a;
#pragma unroll
    for (int i = 0; i < 8; ++i) { a[i] = lo[i]; a[8 + i] = hi[i]; }
    return a;
}
__device__ inline v16bf umamba_load_b(const bf16_t* p /* &col[k_chunk + 16*laneg] */) {
    v8bf lo = *(const v8bf*)p;
    v8bf hi = *(const v8bf*)(p + 8);
    v16bf b;
#pragma unroll
    for (int i = 0; i < 8; ++i) { b[i] = lo[i]; b[8 + i] = hi[i]; }
    return b;
}
#define UMAMBA_WMMA(a, b, c) \
    __builtin_amdgcn_wmma_f32_16x16x32_bf16(false, (a), false, (b), (short)0, (c), false, false)

// ---------------------------------------------------------------------------
// Implicit-GEMM 3x3x3 conv via WMMA.  One wave per 16(M=positions) x 64(N)
// tile: A fragment reused across 4 accumulators.  Padded input removes all
// boundary predication from the inner loop.
//   xt: [B, 26,26,26, Cin] bf16 (zero halo), wt: [27, Cout, Cin] bf16,
//   yt: [B*L, Cout] f32.   grid = (BLM/16, Cout/64)
// ---------------------------------------------------------------------------
__global__ void umamba_conv3d_wmma(const bf16_t* __restrict__ xt,
                                   const bf16_t* __restrict__ wt,
                                   const float*  __restrict__ bias,
                                   float* __restrict__ yt,
                                   int Cin, int Cout) {
    const int mt = blockIdx.x;            // M tile (16 positions)
    const int ng = blockIdx.y;            // group of 4 N tiles (64 out channels)
    const int lane = threadIdx.x;
    const int row = lane & 15, lg = lane >> 4;

    const int m0 = mt * 16;
    const int b  = m0 / LL;
    const int l  = (m0 % LL) + row;       // this lane's position (A row)
    const int z  = l / (HH * HH);
    const int rm = l % (HH * HH);
    const int y0 = rm / HH;
    const int x0 = rm % HH;

    v8f acc0 = {}, acc1 = {}, acc2 = {}, acc3 = {};
    const int ksteps = Cin / 32;
    const bf16_t* bcol = wt + ((long)ng * 64 + row) * Cin;   // n-tile 0 column

    for (int tap = 0; tap < 27; ++tap) {
        const int kz = tap / 9, ky = (tap / 3) % 3, kx = tap % 3;
        const long lpos = (((long)b * HP + (z + kz)) * HP + (y0 + ky)) * HP + (x0 + kx);
        const bf16_t* arow = xt + lpos * Cin;
        const bf16_t* bptr = bcol + (long)tap * Cout * Cin;
        for (int kk = 0; kk < ksteps; ++kk) {
            v16bf aM = umamba_load_a(arow + kk * 32 + 16 * lg);
            const bf16_t* bp = bptr + kk * 32 + 16 * lg;
            v16bf b0 = umamba_load_b(bp);
            v16bf b1 = umamba_load_b(bp + 16 * Cin);
            v16bf b2 = umamba_load_b(bp + 32 * Cin);
            v16bf b3 = umamba_load_b(bp + 48 * Cin);
            acc0 = UMAMBA_WMMA(aM, b0, acc0);
            acc1 = UMAMBA_WMMA(aM, b1, acc1);
            acc2 = UMAMBA_WMMA(aM, b2, acc2);
            acc3 = UMAMBA_WMMA(aM, b3, acc3);
        }
    }
    const int n0 = ng * 64 + row;
    const float bv0 = bias[n0], bv1 = bias[n0 + 16], bv2 = bias[n0 + 32], bv3 = bias[n0 + 48];
#pragma unroll
    for (int v = 0; v < 8; ++v) {
        float* dst = yt + (long)(m0 + 8 * lg + v) * Cout + n0;
        dst[0]  = acc0[v] + bv0;
        dst[16] = acc1[v] + bv1;
        dst[32] = acc2[v] + bv2;
        dst[48] = acc3[v] + bv3;
    }
}

// ---------------------------------------------------------------------------
// GEMM  D[M,N] = A[M,K] * B[N,K]^T  via WMMA; wave computes 16x64 tile.
// grid = (M/16, N/64)
// ---------------------------------------------------------------------------
__global__ void umamba_gemm_wmma(const bf16_t* __restrict__ A,
                                 const bf16_t* __restrict__ Bw,
                                 float* __restrict__ Dst,
                                 int N, int K) {
    const int mt = blockIdx.x, ng = blockIdx.y;
    const int lane = threadIdx.x;
    const int row = lane & 15, lg = lane >> 4;
    const bf16_t* arow = A  + ((long)mt * 16 + row) * K;
    const bf16_t* bcol = Bw + ((long)ng * 64 + row) * K;
    v8f acc0 = {}, acc1 = {}, acc2 = {}, acc3 = {};
    for (int kk = 0; kk < K; kk += 32) {
        v16bf aM = umamba_load_a(arow + kk + 16 * lg);
        const bf16_t* bp = bcol + kk + 16 * lg;
        v16bf b0 = umamba_load_b(bp);
        v16bf b1 = umamba_load_b(bp + 16 * K);
        v16bf b2 = umamba_load_b(bp + 32 * K);
        v16bf b3 = umamba_load_b(bp + 48 * K);
        acc0 = UMAMBA_WMMA(aM, b0, acc0);
        acc1 = UMAMBA_WMMA(aM, b1, acc1);
        acc2 = UMAMBA_WMMA(aM, b2, acc2);
        acc3 = UMAMBA_WMMA(aM, b3, acc3);
    }
    const int n0 = ng * 64 + row;
#pragma unroll
    for (int v = 0; v < 8; ++v) {
        float* dst = Dst + (long)(mt * 16 + 8 * lg + v) * N + n0;
        dst[0]  = acc0[v];
        dst[16] = acc1[v];
        dst[32] = acc2[v];
        dst[48] = acc3[v];
    }
}

// out_proj: yg[BL,128] x Wout[64,128]^T, stored transposed to [B, 64, L] f32
// grid = (BLM/16, 1); wave computes all 64 output channels.
__global__ void umamba_outproj_wmma(const bf16_t* __restrict__ A,
                                    const bf16_t* __restrict__ Bw,
                                    float* __restrict__ out) {
    const int mt = blockIdx.x;
    const int lane = threadIdx.x;
    const int row = lane & 15, lg = lane >> 4;
    const int K = DIN;
    const bf16_t* arow = A  + ((long)mt * 16 + row) * K;
    const bf16_t* bcol = Bw + (long)row * K;
    v8f acc0 = {}, acc1 = {}, acc2 = {}, acc3 = {};
    for (int kk = 0; kk < K; kk += 32) {
        v16bf aM = umamba_load_a(arow + kk + 16 * lg);
        const bf16_t* bp = bcol + kk + 16 * lg;
        v16bf b0 = umamba_load_b(bp);
        v16bf b1 = umamba_load_b(bp + 16 * K);
        v16bf b2 = umamba_load_b(bp + 32 * K);
        v16bf b3 = umamba_load_b(bp + 48 * K);
        acc0 = UMAMBA_WMMA(aM, b0, acc0);
        acc1 = UMAMBA_WMMA(aM, b1, acc1);
        acc2 = UMAMBA_WMMA(aM, b2, acc2);
        acc3 = UMAMBA_WMMA(aM, b3, acc3);
    }
#pragma unroll
    for (int v = 0; v < 8; ++v) {
        long m = (long)mt * 16 + 8 * lg + v;
        int b = (int)(m / LL), l = (int)(m % LL);
        float* dst = out + (long)b * DIMC * LL + l;
        dst[(long)(row)      * LL] = acc0[v];
        dst[(long)(row + 16) * LL] = acc1[v];
        dst[(long)(row + 32) * LL] = acc2[v];
        dst[(long)(row + 48) * LL] = acc3[v];
    }
}

// ---------------------------------------------------------------------------
// InstanceNorm stats per (b,c): mean & rstd over L (layout [b, l, c])
// ---------------------------------------------------------------------------
__global__ void umamba_in_stats(const float* __restrict__ y, int C,
                                float* __restrict__ stats) {
    const int bc = blockIdx.x;
    const int c = bc % C, b = bc / C;
    const float* p = y + (long)b * LL * C + c;
    float s = 0.f, s2 = 0.f;
    for (int l = threadIdx.x; l < LL; l += blockDim.x) {
        float v = p[(long)l * C];
        s += v; s2 += v * v;
    }
    __shared__ float sh[512];
    sh[threadIdx.x] = s; sh[256 + threadIdx.x] = s2;
    __syncthreads();
    for (int off = 128; off > 0; off >>= 1) {
        if ((int)threadIdx.x < off) {
            sh[threadIdx.x] += sh[threadIdx.x + off];
            sh[256 + threadIdx.x] += sh[256 + threadIdx.x + off];
        }
        __syncthreads();
    }
    if (threadIdx.x == 0) {
        float mu  = sh[0] / (float)LL;
        float var = sh[256] / (float)LL - mu * mu;
        stats[bc * 2]     = mu;
        stats[bc * 2 + 1] = rsqrtf(var + 1e-5f);
    }
}

// x1 = IN(y1) + leaky(y1), written as PADDED bf16 [b,26,26,26,c] for conv2
__global__ void umamba_in_apply_pad(const float* __restrict__ y,
                                    const float* __restrict__ stats,
                                    bf16_t* __restrict__ out, int C) {
    long idx = (long)blockIdx.x * blockDim.x + threadIdx.x;
    long n = (long)BB * LP * C;
    if (idx >= n) return;
    int c = idx % C; long r = idx / C;
    int px = r % HP; r /= HP;
    int py = r % HP; r /= HP;
    int pz = r % HP; int b = (int)(r / HP);
    float res = 0.f;
    if (px >= 1 && px < 25 && py >= 1 && py < 25 && pz >= 1 && pz < 25) {
        int l = (pz - 1) * (HH * HH) + (py - 1) * HH + (px - 1);
        float v = y[((long)b * LL + l) * C + c];
        float mu = stats[(b * C + c) * 2], rs = stats[(b * C + c) * 2 + 1];
        res = (v - mu) * rs + (v >= 0.f ? v : 0.01f * v);
    }
    out[idx] = (bf16_t)res;
}

// IN2 + leaky + residual(+inp) + LayerNorm over 64 channels -> seq bf16 [b,l,64]
__global__ void umamba_in2_res_ln(const float* __restrict__ y2,
                                  const float* __restrict__ stats2,
                                  const float* __restrict__ x0,   // [B, 64, L]
                                  const float* __restrict__ lnw,
                                  const float* __restrict__ lnb,
                                  bf16_t* __restrict__ seq) {
    const int pos = blockIdx.x;               // 0..BLM-1
    const int b = pos / LL, l = pos % LL;
    const int c = threadIdx.x;                // 64 threads
    float v = y2[(long)pos * DIMC + c];
    float mu = stats2[(b * DIMC + c) * 2], rs = stats2[(b * DIMC + c) * 2 + 1];
    float t = (v - mu) * rs + (v >= 0.f ? v : 0.01f * v)
              + x0[((long)b * DIMC + c) * LL + l];
    __shared__ float sh[128];
    sh[c] = t; sh[64 + c] = t * t;
    __syncthreads();
    for (int off = 32; off > 0; off >>= 1) {
        if (c < off) { sh[c] += sh[c + off]; sh[64 + c] += sh[64 + c + off]; }
        __syncthreads();
    }
    float mean = sh[0] * (1.f / 64.f);
    float var  = sh[64] * (1.f / 64.f) - mean * mean;
    float r = (t - mean) * rsqrtf(var + 1e-5f) * lnw[c] + lnb[c];
    seq[(long)pos * DIMC + c] = (bf16_t)r;
}

// depthwise causal conv1d + SiLU:  xm[b,l,d] from xz[b,l,0:128]
__global__ void umamba_conv1d_silu(const float* __restrict__ xz,
                                   const float* __restrict__ w,
                                   const float* __restrict__ bias,
                                   float* __restrict__ xm) {
    long idx = (long)blockIdx.x * blockDim.x + threadIdx.x;
    long n = (long)BLM * DIN;
    if (idx >= n) return;
    int d = idx % DIN; long bl = idx / DIN;
    int l = (int)(bl % LL), b = (int)(bl / LL);
    float acc = bias[d];
#pragma unroll
    for (int k = 0; k < 4; ++k) {
        int ll = l - 3 + k;
        if (ll >= 0) acc += w[d * 4 + k] * xz[((long)b * LL + ll) * 256 + d];
    }
    xm[idx] = acc / (1.f + __expf(-acc));      // SiLU
}

// x_proj (36 outputs) + dt_proj + softplus, per position (block = 128 threads)
__global__ void umamba_xproj_dt(const float* __restrict__ xm,
                                const float* __restrict__ xpw,  // [36,128]
                                const float* __restrict__ dtw,  // [128,4]
                                const float* __restrict__ dtb,
                                float* __restrict__ delta,
                                float* __restrict__ Bm,
                                float* __restrict__ Cm) {
    const int pos = blockIdx.x;
    const int t = threadIdx.x;                 // 128
    __shared__ float shx[128];
    __shared__ float shd[36];
    shx[t] = xm[(long)pos * DIN + t];
    __syncthreads();
    if (t < 36) {
        float acc = 0.f;
#pragma unroll 8
        for (int k = 0; k < 128; ++k) acc += xpw[t * 128 + k] * shx[k];
        shd[t] = acc;
        if (t >= 4 && t < 20)      Bm[(long)pos * NST + (t - 4)]  = acc;
        else if (t >= 20)          Cm[(long)pos * NST + (t - 20)] = acc;
    }
    __syncthreads();
    float acc = dtb[t];
#pragma unroll
    for (int r = 0; r < 4; ++r) acc += dtw[t * 4 + r] * shd[r];
    float sp = (acc > 20.f) ? acc : __logf(1.f + __expf(acc));   // softplus
    delta[(long)pos * DIN + t] = sp;
}

// ---------------------------------------------------------------------------
// Selective scan: one wave per 2 channels; lane = (laneg: which d) * 16 + n.
// State s[d,n] lives in a register; y_t reduced over n with 16-lane shfl_xor.
// Fuses  y = (scan + xm*D) * silu(res)  and stores bf16 for out_proj WMMA.
// ---------------------------------------------------------------------------
__global__ void umamba_scan(const float* __restrict__ delta,
                            const float* __restrict__ xm,
                            const float* __restrict__ Bm,
                            const float* __restrict__ Cm,
                            const float* __restrict__ A_log,
                            const float* __restrict__ Dp,
                            const float* __restrict__ xz,    // res = xz[..,128+d]
                            bf16_t* __restrict__ yg) {
    const int wid = blockIdx.x;               // 0 .. B*64-1
    const int b = wid / 64, dpair = wid % 64;
    const int lane = threadIdx.x;
    const int n = lane & 15, lg = lane >> 4;
    const int d = dpair * 2 + lg;

    const float A  = -__expf(A_log[d * NST + n]);
    const float Dv = Dp[d];
    float s = 0.f;

    const float* dl = delta + (long)b * LL * DIN + d;
    const float* xl = xm    + (long)b * LL * DIN + d;
    const float* Bl = Bm    + (long)b * LL * NST + n;
    const float* Cl = Cm    + (long)b * LL * NST + n;
    const float* rl = xz    + (long)b * LL * 256 + 128 + d;
    bf16_t*      yl = yg    + (long)b * LL * DIN + d;

    for (int l = 0; l < LL; ++l) {
        float dt = dl[(long)l * DIN];
        float xv = xl[(long)l * DIN];
        float bv = Bl[(long)l * NST];
        float cv = Cl[(long)l * NST];
        s = __expf(dt * A) * s + dt * bv * xv;
        float contrib = s * cv;
        contrib += __shfl_xor(contrib, 1, 32);
        contrib += __shfl_xor(contrib, 2, 32);
        contrib += __shfl_xor(contrib, 4, 32);
        contrib += __shfl_xor(contrib, 8, 32);
        if (n == 0) {
            float res = rl[(long)l * 256];
            float yv = (contrib + xv * Dv) * (res / (1.f + __expf(-res)));
            yl[(long)l * DIN] = (bf16_t)yv;
        }
    }
}

// ---------------------------------------------------------------------------
extern "C" void kernel_launch(void* const* d_in, const int* in_sizes, int n_in,
                              void* d_out, int out_size, void* d_ws, size_t ws_size,
                              hipStream_t stream) {
    const float* x        = (const float*)d_in[0];
    const float* conv1_w  = (const float*)d_in[1];
    const float* conv1_b  = (const float*)d_in[2];
    const float* conv2_w  = (const float*)d_in[3];
    const float* conv2_b  = (const float*)d_in[4];
    const float* ln_w     = (const float*)d_in[5];
    const float* ln_b     = (const float*)d_in[6];
    const float* in_proj  = (const float*)d_in[7];
    const float* c1d_w    = (const float*)d_in[8];
    const float* c1d_b    = (const float*)d_in[9];
    const float* xproj_w  = (const float*)d_in[10];
    const float* dt_w     = (const float*)d_in[11];
    const float* dt_b     = (const float*)d_in[12];
    const float* A_log    = (const float*)d_in[13];
    const float* D_param  = (const float*)d_in[14];
    const float* outp_w   = (const float*)d_in[15];
    float* out = (float*)d_out;

    char* ws = (char*)d_ws;
    auto carve = [&](size_t bytes) {
        char* p = ws;
        ws += (bytes + 255) & ~(size_t)255;
        return p;
    };
    bf16_t* xt     = (bf16_t*)carve((size_t)BB * LP * DIMC * 2);   // padded input bf16
    bf16_t* w1t    = (bf16_t*)carve((size_t)27 * DIN * DIMC * 2);  // [tap][o][i]
    bf16_t* w2t    = (bf16_t*)carve((size_t)27 * DIMC * DIN * 2);
    bf16_t* wip    = (bf16_t*)carve((size_t)256 * DIMC * 2);
    bf16_t* wop    = (bf16_t*)carve((size_t)DIMC * DIN * 2);
    float*  y1f    = (float*) carve((size_t)BLM * DIN * 4);
    float*  stats1 = (float*) carve((size_t)BB * DIN * 2 * 4);
    bf16_t* x1t    = (bf16_t*)carve((size_t)BB * LP * DIN * 2);    // padded conv2 input
    float*  y2f    = (float*) carve((size_t)BLM * DIMC * 4);
    float*  stats2 = (float*) carve((size_t)BB * DIMC * 2 * 4);
    bf16_t* seq    = (bf16_t*)carve((size_t)BLM * DIMC * 2);
    float*  xz     = (float*) carve((size_t)BLM * 256 * 4);
    float*  xmv    = (float*) carve((size_t)BLM * DIN * 4);
    float*  deltav = (float*) carve((size_t)BLM * DIN * 4);
    float*  Bmv    = (float*) carve((size_t)BLM * NST * 4);
    float*  Cmv    = (float*) carve((size_t)BLM * NST * 4);
    bf16_t* yg     = (bf16_t*)carve((size_t)BLM * DIN * 2);
    (void)ws_size; (void)in_sizes; (void)n_in; (void)out_size;

    const int TPB = 256;
    auto blocks = [](long n, int t) { return (unsigned)((n + t - 1) / t); };

    // 1) repack / bf16 conversion (padded layouts for the convs)
    umamba_pad_transpose<<<blocks((long)BB * LP * DIMC, TPB), TPB, 0, stream>>>(x, xt);
    umamba_repack_convw<<<blocks((long)DIN * DIMC * 27, TPB), TPB, 0, stream>>>(conv1_w, w1t, DIN, DIMC);
    umamba_repack_convw<<<blocks((long)DIMC * DIN * 27, TPB), TPB, 0, stream>>>(conv2_w, w2t, DIMC, DIN);
    umamba_f2bf<<<blocks(256L * DIMC, TPB), TPB, 0, stream>>>(in_proj, wip, 256L * DIMC);
    umamba_f2bf<<<blocks((long)DIMC * DIN, TPB), TPB, 0, stream>>>(outp_w, wop, (long)DIMC * DIN);

    // 2) conv1 (WMMA, 16x64 tile/wave) -> IN stats -> IN+leaky (padded bf16)
    umamba_conv3d_wmma<<<dim3(BLM / 16, DIN / 64), 32, 0, stream>>>(xt, w1t, conv1_b, y1f, DIMC, DIN);
    umamba_in_stats<<<BB * DIN, 256, 0, stream>>>(y1f, DIN, stats1);
    umamba_in_apply_pad<<<blocks((long)BB * LP * DIN, TPB), TPB, 0, stream>>>(y1f, stats1, x1t, DIN);

    // 3) conv2 (WMMA) -> IN stats -> fused IN+leaky+residual+LN (bf16 seq)
    umamba_conv3d_wmma<<<dim3(BLM / 16, DIMC / 64), 32, 0, stream>>>(x1t, w2t, conv2_b, y2f, DIN, DIMC);
    umamba_in_stats<<<BB * DIMC, 256, 0, stream>>>(y2f, DIMC, stats2);
    umamba_in2_res_ln<<<BLM, DIMC, 0, stream>>>(y2f, stats2, x, ln_w, ln_b, seq);

    // 4) in_proj (WMMA): [BL,64] x [256,64]^T -> xz [BL,256]
    umamba_gemm_wmma<<<dim3(BLM / 16, 256 / 64), 32, 0, stream>>>(seq, wip, xz, 256, DIMC);

    // 5) causal dw-conv1d + SiLU
    umamba_conv1d_silu<<<blocks((long)BLM * DIN, TPB), TPB, 0, stream>>>(xz, c1d_w, c1d_b, xmv);

    // 6) x_proj + dt_proj + softplus (delta, B, C)
    umamba_xproj_dt<<<BLM, 128, 0, stream>>>(xmv, xproj_w, dt_w, dt_b, deltav, Bmv, Cmv);

    // 7) selective scan (register-resident state, shfl reduction), fused gate
    umamba_scan<<<BB * 64, 32, 0, stream>>>(deltav, xmv, Bmv, Cmv, A_log, D_param, xz, yg);

    // 8) out_proj (WMMA) with transposed store to [B, 64, H, W, D]
    umamba_outproj_wmma<<<dim3(BLM / 16, 1), 32, 0, stream>>>(yg, wop, out);
}
